// SpikeDrivenSelfAttention_8272107012220
// MI455X (gfx1250) — 16-bit and 8-bit matrix operand order repaired and checked by reading, compiled, not run on hardware
//
#include <hip/hip_runtime.h>
#include <hip/hip_bf16.h>
#include <math.h>


typedef __attribute__((ext_vector_type(8)))  __bf16 v8bf;
typedef __attribute__((ext_vector_type(16))) __bf16 v16bf;
typedef __attribute__((ext_vector_type(8)))  float  v8f;
typedef __attribute__((ext_vector_type(8)))  int    v8i;

constexpr int Bsz  = 2;
constexpr int Nseq = 2048;
constexpr int Cdim = 1024;
constexpr int Hh   = 16;
constexpr int Dd   = 64;
constexpr int Mtot = Bsz * Nseq;

constexpr size_t OFF_XB   = 0;
constexpr size_t OFF_WQ   = OFF_XB   + (size_t)Mtot * Cdim * 2;
constexpr size_t OFF_WK   = OFF_WQ   + (size_t)Cdim * Cdim * 2;
constexpr size_t OFF_WV   = OFF_WK   + (size_t)Cdim * Cdim * 2;
constexpr size_t OFF_WO   = OFF_WV   + (size_t)Cdim * Cdim * 2;
constexpr size_t OFF_Q8   = OFF_WO   + (size_t)Cdim * Cdim * 2;
constexpr size_t OFF_K8   = OFF_Q8   + (size_t)Mtot * Cdim;
constexpr size_t OFF_VT   = OFF_K8   + (size_t)Mtot * Cdim;
constexpr size_t OFF_ATT  = OFF_VT   + (size_t)Cdim * Mtot * 2;


static __device__ __forceinline__ unsigned short f2bf(float x) {
    unsigned u = __builtin_bit_cast(unsigned, x);
    unsigned r = u + 0x7FFFu + ((u >> 16) & 1u);
    return (unsigned short)(r >> 16);
}

static __device__ __forceinline__ float fsig(float x) {
    return 1.0f / (1.0f + __expf(-x));
}

static __device__ __forceinline__ void async_copy_b128(unsigned ldsOff, const void* gsrc) {
    asm volatile("global_load_async_to_lds_b128 %0, %1, off"
                 :: "v"(ldsOff), "v"(gsrc)
                 : "memory");
}

static __device__ __forceinline__ void wait_async0() {
#if __has_builtin(__builtin_amdgcn_s_wait_asynccnt)
    __builtin_amdgcn_s_wait_asynccnt(0);
#else
    asm volatile("s_wait_asynccnt 0x0" ::: "memory");
#endif
}

static __device__ __forceinline__ unsigned lds_off_u32(const void* p) {
    return (unsigned)(unsigned long long)p;
}

static __device__ __forceinline__ v16bf load_fragA_bf16(const unsigned short* row, int hs) {
    const v8bf lo = *reinterpret_cast<const v8bf*>(row + 8 * hs);
    const v8bf hi = *reinterpret_cast<const v8bf*>(row + 16 + 8 * hs);
    v16bf f;
#pragma unroll
    for (int i = 0; i < 8; ++i) { f[i] = lo[i]; f[8 + i] = hi[i]; }
    return f;
}

static __device__ __forceinline__ v16bf load_fragB_bf16(const unsigned short* row, int hs) {
    const v8bf lo = *reinterpret_cast<const v8bf*>(row + 8 * hs);
    const v8bf hi = *reinterpret_cast<const v8bf*>(row + 16 + 8 * hs);
    v16bf f;
#pragma unroll
    for (int i = 0; i < 8; ++i) { f[i] = lo[i]; f[8 + i] = hi[i]; }
    return f;
}

static __device__ __forceinline__ v8i load_fragA_iu8(const unsigned char* row, int hs) {
    const int2 c0 = *reinterpret_cast<const int2*>(row +      8 * hs);
    const int2 c1 = *reinterpret_cast<const int2*>(row + 16 + 8 * hs);
    const int2 c2 = *reinterpret_cast<const int2*>(row + 32 + 8 * hs);
    const int2 c3 = *reinterpret_cast<const int2*>(row + 48 + 8 * hs);
    v8i f;
    f[0] = c0.x; f[1] = c0.y; f[2] = c1.x; f[3] = c1.y;
    f[4] = c2.x; f[5] = c2.y; f[6] = c3.x; f[7] = c3.y;
    return f;
}

static __device__ __forceinline__ v8i load_fragB_iu8(const unsigned char* row, int hs) {
    const int2 c0 = *reinterpret_cast<const int2*>(row +      8 * hs);
    const int2 c1 = *reinterpret_cast<const int2*>(row + 16 + 8 * hs);
    const int2 c2 = *reinterpret_cast<const int2*>(row + 32 + 8 * hs);
    const int2 c3 = *reinterpret_cast<const int2*>(row + 48 + 8 * hs);
    v8i f = { c0.x, c0.y, c1.x, c1.y, c2.x, c2.y, c3.x, c3.y };
    return f;
}


__global__ void cvt_f32_bf16(const float* __restrict__ in,
                             unsigned short* __restrict__ out, int n) {
    int i = blockIdx.x * 256 + threadIdx.x;
    if (i < n) out[i] = f2bf(in[i]);
}

template <int MODE>
__global__ __launch_bounds__(256) void gemm_wmma_bf16(
    const unsigned short* __restrict__ Xb,
    const unsigned short* __restrict__ Wb,
    const float* __restrict__ bias,
    const float* __restrict__ th,
    unsigned char*  __restrict__ out8,
    unsigned short* __restrict__ outT,
    float*          __restrict__ outF)
{
    constexpr int K = Cdim;
    __shared__ __align__(16) unsigned short sX[128][32];
    __shared__ __align__(16) unsigned short sW[64][32];

    const int tid  = threadIdx.x;
    const int lane = tid & 31;
    const int wid  = tid >> 5;
    const int hs   = lane >> 4;
    const int ln   = lane & 15;
    const int m_loc = (wid & 3) * 32;
    const int n_loc = (wid >> 2) * 32;
    const int m_blk = blockIdx.y * 128;
    const int n_blk = blockIdx.x * 64;

    v8f acc[2][2];
#pragma unroll
    for (int i = 0; i < 2; ++i)
#pragma unroll
        for (int j = 0; j < 2; ++j)
#pragma unroll
            for (int e = 0; e < 8; ++e) acc[i][j][e] = 0.0f;

    for (int k0 = 0; k0 < K; k0 += 32) {
        __syncthreads();

#pragma unroll
        for (int p = 0; p < 2; ++p) {
            const int c   = tid + 256 * p;
            const int row = c >> 2;
            const int col = (c & 3) * 8;
            async_copy_b128(lds_off_u32(&sX[row][col]),
                            Xb + (size_t)(m_blk + row) * K + k0 + col);
        }
        {
            const int row = tid >> 2;
            const int col = (tid & 3) * 8;
            async_copy_b128(lds_off_u32(&sW[row][col]),
                            Wb + (size_t)(n_blk + row) * K + k0 + col);
        }
        wait_async0();
        __syncthreads();

        v16bf afr[2], bfr[2];
#pragma unroll
        for (int i = 0; i < 2; ++i)
            afr[i] = load_fragA_bf16(&sX[m_loc + i * 16 + ln][0], hs);
#pragma unroll
        for (int j = 0; j < 2; ++j)
            bfr[j] = load_fragB_bf16(&sW[n_loc + j * 16 + ln][0], hs);

#pragma unroll
        for (int i = 0; i < 2; ++i)
#pragma unroll
            for (int j = 0; j < 2; ++j)
                acc[i][j] = __builtin_amdgcn_wmma_f32_16x16x32_bf16(
                    false, afr[i], false, bfr[j], (short)0, acc[i][j], false, false);
    }

#pragma unroll
    for (int i = 0; i < 2; ++i)
#pragma unroll
        for (int j = 0; j < 2; ++j) {
            const int n = n_blk + n_loc + j * 16 + ln;
            const float bn = bias[n];
#pragma unroll
            for (int r = 0; r < 8; ++r) {
                const int m = m_blk + m_loc + i * 16 + r + 8 * hs;
                const float val = acc[i][j][r] + bn;
                if (MODE == 0) {
                    out8[(size_t)m * Cdim + n] = (val > th[n]) ? (unsigned char)1 : (unsigned char)0;
                } else if (MODE == 1) {
                    outT[(size_t)n * Mtot + m] = f2bf(val);
                } else {
                    outF[(size_t)m * Cdim + n] = val;
                }
            }
        }
}

__global__ __launch_bounds__(128) void attn_wmma(
    const unsigned char*  __restrict__ q8,
    const unsigned char*  __restrict__ k8,
    const unsigned short* __restrict__ vT,
    unsigned short*       __restrict__ attn)
{
    __shared__ __align__(16) unsigned short wbuf[4][16][32];

    const int lane = threadIdx.x & 31;
    const int wid  = threadIdx.x >> 5;
    const int hs   = lane >> 4;
    const int ln   = lane & 15;
    const int bh   = blockIdx.y;
    const int b    = bh >> 4;
    const int h    = bh & 15;
    const int m0   = (blockIdx.x * 4 + wid) * 16;
    const size_t tokBase = (size_t)b * Nseq;
    const float scale = 0.125f;

    const v8i aq = load_fragA_iu8(q8 + (tokBase + m0 + ln) * Cdim + h * Dd, hs);
    const v8i zi = { 0, 0, 0, 0, 0, 0, 0, 0 };

    v8f acc[4];
#pragma unroll
    for (int t = 0; t < 4; ++t)
#pragma unroll
        for (int e = 0; e < 8; ++e) acc[t][e] = 0.0f;

    for (int n0 = 0; n0 < Nseq; n0 += 32) {
        const v8i bk0 = load_fragB_iu8(k8 + (tokBase + n0 +      ln) * Cdim + h * Dd, hs);
        const v8i bk1 = load_fragB_iu8(k8 + (tokBase + n0 + 16 + ln) * Cdim + h * Dd, hs);

        const v8i s0 = __builtin_amdgcn_wmma_i32_16x16x64_iu8(false, aq, false, bk0, zi, false, false);
        const v8i s1 = __builtin_amdgcn_wmma_i32_16x16x64_iu8(false, aq, false, bk1, zi, false, false);

#pragma unroll
        for (int r = 0; r < 8; ++r) {
            const int mm = r + 8 * hs;
            wbuf[wid][mm][ln]      = f2bf(fsig((float)s0[r] * scale));
            wbuf[wid][mm][16 + ln] = f2bf(fsig((float)s1[r] * scale));
        }

        const v16bf aw = load_fragA_bf16(&wbuf[wid][ln][0], hs);

#pragma unroll
        for (int t = 0; t < 4; ++t) {
            const unsigned short* vb = vT + (size_t)(h * Dd + t * 16 + ln) * Mtot + tokBase + n0;
            const v16bf bv = load_fragB_bf16(vb, hs);
            acc[t] = __builtin_amdgcn_wmma_f32_16x16x32_bf16(
                false, aw, false, bv, (short)0, acc[t], false, false);
        }
    }

#pragma unroll
    for (int t = 0; t < 4; ++t)
#pragma unroll
        for (int r = 0; r < 8; ++r) {
            const int m  = m0 + r + 8 * hs;
            const int ch = h * Dd + t * 16 + ln;
            attn[(tokBase + m) * Cdim + ch] = f2bf(acc[t][r]);
        }
}


extern "C" void kernel_launch(void* const* d_in, const int* in_sizes, int n_in,
                              void* d_out, int out_size, void* d_ws, size_t ws_size,
                              hipStream_t stream) {
    (void)in_sizes; (void)n_in; (void)out_size; (void)ws_size;

    const float* x    = (const float*)d_in[0];
    const float* Wq   = (const float*)d_in[1];
    const float* bq   = (const float*)d_in[2];
    const float* Wk   = (const float*)d_in[3];
    const float* bk   = (const float*)d_in[4];
    const float* Wv   = (const float*)d_in[5];
    const float* bv   = (const float*)d_in[6];
    const float* Wo   = (const float*)d_in[7];
    const float* bo   = (const float*)d_in[8];
    const float* th_q = (const float*)d_in[9];
    const float* th_k = (const float*)d_in[10];
    float* out = (float*)d_out;

    char* ws = (char*)d_ws;
    unsigned short* xb   = (unsigned short*)(ws + OFF_XB);
    unsigned short* wqb  = (unsigned short*)(ws + OFF_WQ);
    unsigned short* wkb  = (unsigned short*)(ws + OFF_WK);
    unsigned short* wvb  = (unsigned short*)(ws + OFF_WV);
    unsigned short* wob  = (unsigned short*)(ws + OFF_WO);
    unsigned char*  q8   = (unsigned char*) (ws + OFF_Q8);
    unsigned char*  k8   = (unsigned char*) (ws + OFF_K8);
    unsigned short* vT   = (unsigned short*)(ws + OFF_VT);
    unsigned short* attn = (unsigned short*)(ws + OFF_ATT);

    {
        const int nx = Mtot * Cdim;
        const int nw = Cdim * Cdim;
        cvt_f32_bf16<<<nx / 256, 256, 0, stream>>>(x,  xb,  nx);
        cvt_f32_bf16<<<nw / 256, 256, 0, stream>>>(Wq, wqb, nw);
        cvt_f32_bf16<<<nw / 256, 256, 0, stream>>>(Wk, wkb, nw);
        cvt_f32_bf16<<<nw / 256, 256, 0, stream>>>(Wv, wvb, nw);
        cvt_f32_bf16<<<nw / 256, 256, 0, stream>>>(Wo, wob, nw);
    }

    const dim3 ggrid(Cdim / 64, Mtot / 128);
    const dim3 gblk(256);

    gemm_wmma_bf16<0><<<ggrid, gblk, 0, stream>>>(xb, wqb, bq, th_q, q8, nullptr, nullptr);
    gemm_wmma_bf16<0><<<ggrid, gblk, 0, stream>>>(xb, wkb, bk, th_k, k8, nullptr, nullptr);
    gemm_wmma_bf16<1><<<ggrid, gblk, 0, stream>>>(xb, wvb, bv, nullptr, nullptr, vT, nullptr);

    attn_wmma<<<dim3(Nseq / 64, Bsz * Hh), dim3(128), 0, stream>>>(q8, k8, vT, attn);

    gemm_wmma_bf16<2><<<ggrid, gblk, 0, stream>>>(attn, wob, bo, nullptr, nullptr, nullptr, out);
}
